// EnhancedGNNModelHousehold_20349555049091
// MI455X (gfx1250) — matrix operand order repaired and checked by reading, not compiled, not run on hardware
//
#include <hip/hip_runtime.h>
#include <stdint.h>

// ---------------------------------------------------------------------------
// CDNA5 (gfx1250) fused GNN pipeline, bf16 activations + v_wmma_f32_16x16x32_bf16
// Async global->LDS staging (ASYNCcnt) when the toolchain exposes the builtin.
// ---------------------------------------------------------------------------

typedef __attribute__((ext_vector_type(16))) __bf16 bf16x16;
typedef __attribute__((ext_vector_type(8)))  float  f32x8;

union ABFrag { uint32_t u[8]; bf16x16 v; };

#if defined(__has_builtin)
#if __has_builtin(__builtin_amdgcn_global_load_async_to_lds_b128)
#define USE_ASYNC_LDS 1
#endif
#endif

#ifdef USE_ASYNC_LDS
typedef int v4i __attribute__((ext_vector_type(4)));
typedef __attribute__((address_space(1))) v4i g_v4i;   // global
typedef __attribute__((address_space(3))) v4i l_v4i;   // LDS
__device__ __forceinline__ void async_copy16(const void* g, void* l) {
  __builtin_amdgcn_global_load_async_to_lds_b128((g_v4i*)(void*)g, (l_v4i*)l, 0, 0);
}
__device__ __forceinline__ void wait_async0() {
#if __has_builtin(__builtin_amdgcn_s_wait_asynccnt)
  __builtin_amdgcn_s_wait_asynccnt(0);
#else
  asm volatile("s_wait_asynccnt 0" ::: "memory");
#endif
}
#endif

__device__ __forceinline__ uint16_t f2b(float f) {
  uint32_t u = __float_as_uint(f);
  u += 0x7FFFu + ((u >> 16) & 1u);          // round-to-nearest-even
  return (uint16_t)(u >> 16);
}
__device__ __forceinline__ float b2f(uint16_t h) {
  return __uint_as_float(((uint32_t)h) << 16);
}
__device__ __forceinline__ uint32_t pack2(float lo, float hi) {
  return (uint32_t)f2b(lo) | ((uint32_t)f2b(hi) << 16);
}

// A fragment: 16x32 bf16, rows m0..m0+15, K chunk kc (K=32*kc..+31).
// Row-major bf16 packed as u32 pairs, strideU u32 per row.
__device__ __forceinline__ bf16x16 ldA(const uint32_t* p, int m0, int kc, int strideU) {
  int lane = threadIdx.x & 31;
  const uint32_t* r = p + (size_t)(m0 + (lane & 15)) * strideU + kc * 16 + (lane >> 4) * 4;
  ABFrag f;
#pragma unroll
  for (int i = 0; i < 4; ++i) { f.u[i] = r[i]; f.u[4 + i] = r[8 + i]; }
  return f.v;
}

// B fragment: 32x16 bf16 from *transposed* weights wT[n][k] (strideU u32/row).
__device__ __forceinline__ bf16x16 ldB(const uint32_t* wT, int n0, int kc, int strideU) {
  int lane = threadIdx.x & 31;
  const uint32_t* r = wT + (size_t)(n0 + (lane & 15)) * strideU + kc * 16 + (lane >> 4) * 4;
  ABFrag f;
#pragma unroll
  for (int i = 0; i < 4; ++i) { f.u[i] = r[i]; f.u[4 + i] = r[8 + i]; }
  return f.v;
}

__device__ __forceinline__ f32x8 wmma_bf16(bf16x16 a, bf16x16 b, f32x8 c) {
  return __builtin_amdgcn_wmma_f32_16x16x32_bf16(false, a, false, b, (short)0, c,
                                                 false, false);
}

// ---------------------------------------------------------------------------
// Weight prep: f32 [K][Nc] row-major  ->  bf16 transposed [Np][K] (zero pad)
// ---------------------------------------------------------------------------
__global__ void k_prepT(uint16_t* __restrict__ dstW, const float* __restrict__ src,
                        int K, int Nc, int Np) {
  int total = Np * K;
  for (int i = blockIdx.x * 256 + threadIdx.x; i < total; i += gridDim.x * 256) {
    int n = i / K, k = i % K;
    float v = (n < Nc) ? src[(size_t)k * Nc + n] : 0.f;
    dstW[(size_t)n * K + k] = f2b(v);
  }
}

// ---------------------------------------------------------------------------
// Edge counts (per hub) + layer-1 scalar aggregation (x is N x 1)
// ---------------------------------------------------------------------------
__global__ __launch_bounds__(256) void k_count_agg1(const int* __restrict__ ei,
                                                    const float* __restrict__ x,
                                                    float* __restrict__ cnt,
                                                    float* __restrict__ agg1, int H) {
  __shared__ float cL[27], aL[27];
  int tid = threadIdx.x;
  if (tid < 27) { cL[tid] = 0.f; aL[tid] = 0.f; }
  __syncthreads();
  int E = 2 * H;
  for (int e = blockIdx.x * 256 + tid; e < E; e += gridDim.x * 256) {
    int d = ei[E + e] - H;   // dst bin in [0,27)
    int s = ei[e];           // src node (household)
    atomicAdd(&cL[d], 1.f);
    atomicAdd(&aL[d], x[s]);
  }
  __syncthreads();
  if (tid < 27) { atomicAdd(&cnt[tid], cL[tid]); atomicAdd(&agg1[tid], aL[tid]); }
}

// ---------------------------------------------------------------------------
// Layer 1: z1[i,:] = x[i]*w1r + b1 (outer product), + column stats
// ---------------------------------------------------------------------------
__global__ __launch_bounds__(128) void k_layer1(const float* __restrict__ x,
                                                const float* __restrict__ w1r,
                                                const float* __restrict__ b1,
                                                uint16_t* __restrict__ z,
                                                float* __restrict__ sums, int N) {
  __shared__ float wl[128], bl[128];
  int tid = threadIdx.x;
  wl[tid] = w1r[tid]; bl[tid] = b1[tid];
  __syncthreads();
  int r0 = blockIdx.x * 512;
  float wc = wl[tid], bc = bl[tid];
  float s = 0.f, q = 0.f;
  int rmax = min(512, N - r0);
  for (int r = 0; r < rmax; ++r) {
    int g = r0 + r;
    float val = fmaf(x[g], wc, bc);
    z[(size_t)g * 128 + tid] = f2b(val);
    s += val; q += val * val;
  }
  atomicAdd(&sums[tid], s);
  atomicAdd(&sums[128 + tid], q);
}

// ---------------------------------------------------------------------------
// Hub fixup: mean = agg/cnt (27 x K); z[H+j,:] += mean[j,:] @ wl ; patch stats
// ---------------------------------------------------------------------------
__global__ __launch_bounds__(256) void k_fixup(const float* __restrict__ agg,
                                               const float* __restrict__ cnt,
                                               const float* __restrict__ wl, int K,
                                               uint16_t* __restrict__ z,
                                               float* __restrict__ sums, int H) {
  __shared__ float mean[27 * 128];
  int tid = threadIdx.x;
  for (int i = tid; i < 27 * K; i += 256) {
    int j = i / K;
    mean[i] = agg[i] / fmaxf(cnt[j], 1.f);
  }
  __syncthreads();
  for (int i = tid; i < 27 * 128; i += 256) {
    int j = i >> 7, c = i & 127;
    float d = 0.f;
    for (int k = 0; k < K; ++k) d = fmaf(mean[j * K + k], wl[(size_t)k * 128 + c], d);
    size_t idx = (size_t)(H + j) * 128 + c;
    float oldv = b2f(z[idx]);
    float newv = oldv + d;
    z[idx] = f2b(newv);
    atomicAdd(&sums[c], newv - oldv);
    atomicAdd(&sums[128 + c], newv * newv - oldv * oldv);
  }
}

// ---------------------------------------------------------------------------
// Stats finalize: fold graph-norm into per-column affine relu(a*z + c)
// ---------------------------------------------------------------------------
__global__ void k_stats(const float* __restrict__ sums, const float* __restrict__ gw,
                        const float* __restrict__ gb, const float* __restrict__ gm,
                        float* __restrict__ ac, float Ninv) {
  int c = threadIdx.x;
  float mu  = sums[c] * Ninv;
  float ex2 = sums[128 + c] * Ninv;
  float ms  = gm[c];
  float var = ex2 - mu * mu * (2.f * ms - ms * ms);
  float a   = gw[c] * rsqrtf(var + 1e-5f);
  ac[c]       = a;
  ac[128 + c] = gb[c] - a * ms * mu;
}

// ---------------------------------------------------------------------------
// Fused SAGE layer (layers 2,3): per 128-row tile
//   h = relu(a*z_prev + c)  (async-staged to LDS, transformed in place)
//   z = h @ wrT + b         (WMMA bf16 -> f32, written as bf16)
//   column sum / sumsq of z, 27x128 hub bins of h
// ---------------------------------------------------------------------------
__global__ __launch_bounds__(256) void k_ml(const uint16_t* __restrict__ zin,
                                            const float* __restrict__ ac,
                                            const uint32_t* __restrict__ wT,  // [128][64] u32
                                            const float* __restrict__ bias,
                                            uint16_t* __restrict__ zout,
                                            float* __restrict__ sums,
                                            float* __restrict__ agg,
                                            const int* __restrict__ dst,  // ei + 2H
                                            int N, int H) {
  __shared__ uint32_t hT[128 * 64];   // 128 rows x 128 bf16 cols
  __shared__ float bins[27 * 128];
  __shared__ float lsum[128], lsq[128];
  __shared__ int   binE[128], binR[128];
  __shared__ float aL[128], cL[128], bL[128];

  int tid = threadIdx.x;
  int r0  = blockIdx.x * 128;
  bool fullTile = (r0 + 128 <= N);

  for (int i = tid; i < 27 * 128; i += 256) bins[i] = 0.f;
  if (tid < 128) {
    lsum[tid] = 0.f; lsq[tid] = 0.f;
    aL[tid] = ac[tid]; cL[tid] = ac[128 + tid]; bL[tid] = bias[tid];
    int g = r0 + tid;
    if (g < H) { binE[tid] = dst[g] - H; binR[tid] = dst[H + g] - H; }
    else       { binE[tid] = -1;         binR[tid] = -1; }
  }
  __syncthreads();

#ifdef USE_ASYNC_LDS
  // async copy raw z tile (16B chunks, 16 per row), then transform in place
  for (int ch = tid; ch < 128 * 16; ch += 256) {
    int g = r0 + (ch >> 4);
    if (g < N)
      async_copy16((const char*)zin + (size_t)g * 256 + (size_t)(ch & 15) * 16,
                   (char*)hT + (size_t)ch * 16);
  }
  wait_async0();
  __syncthreads();
  for (int idx = tid; idx < 128 * 64; idx += 256) {
    int row = idx >> 6, cp = idx & 63, g = r0 + row;
    uint32_t u = 0;
    if (g < N) {
      uint32_t zp = hT[idx];
      int c0 = cp * 2;
      float h0 = fmaxf(fmaf(aL[c0],     b2f((uint16_t)zp),         cL[c0]),     0.f);
      float h1 = fmaxf(fmaf(aL[c0 + 1], b2f((uint16_t)(zp >> 16)), cL[c0 + 1]), 0.f);
      u = pack2(h0, h1);
    }
    hT[idx] = u;
  }
#else
  const uint32_t* zin32 = (const uint32_t*)zin;
  for (int idx = tid; idx < 128 * 64; idx += 256) {
    int row = idx >> 6, cp = idx & 63, g = r0 + row;
    uint32_t u = 0;
    if (g < N) {
      uint32_t zp = zin32[(size_t)g * 64 + cp];
      int c0 = cp * 2;
      float h0 = fmaxf(fmaf(aL[c0],     b2f((uint16_t)zp),         cL[c0]),     0.f);
      float h1 = fmaxf(fmaf(aL[c0 + 1], b2f((uint16_t)(zp >> 16)), cL[c0 + 1]), 0.f);
      u = pack2(h0, h1);
    }
    hT[idx] = u;
  }
#endif
  __syncthreads();

  int w = tid >> 5, lane = tid & 31, hi = lane >> 4, nl = lane & 15;
  int m0 = w * 16;

  ABFrag a4[4];
#pragma unroll
  for (int kc = 0; kc < 4; ++kc) a4[kc].v = ldA(hT, m0, kc, 64);

#pragma unroll
  for (int nt = 0; nt < 8; ++nt) {
    f32x8 acc = {};
#pragma unroll
    for (int kc = 0; kc < 4; ++kc) {
      bf16x16 b = ldB(wT, nt * 16, kc, 64);
      acc = wmma_bf16(a4[kc].v, b, acc);
    }
    int col = nt * 16 + nl;
    float bcol = bL[col];
    float s = 0.f, q = 0.f;
    uint16_t* zo = zout + (size_t)(r0 + m0 + hi * 8) * 128 + col;
    if (fullTile) {
#pragma unroll
      for (int v = 0; v < 8; ++v) {
        float val = acc[v] + bcol;
        zo[(size_t)v * 128] = f2b(val);
        s += val; q += val * val;
      }
    } else {
#pragma unroll
      for (int v = 0; v < 8; ++v) {
        int g = r0 + m0 + v + hi * 8;
        if (g < N) {
          float val = acc[v] + bcol;
          zo[(size_t)v * 128] = f2b(val);
          s += val; q += val * val;
        }
      }
    }
    atomicAdd(&lsum[col], s);
    atomicAdd(&lsq[col], q);
  }

  // hub binning from h tile (household rows only)
  {
    int col = tid & 127, rs = tid >> 7;
    for (int row = rs; row < 128; row += 2) {
      int be = binE[row];
      if (be >= 0) {
        uint32_t zp = hT[row * 64 + (col >> 1)];
        float hv = b2f((uint16_t)(zp >> ((col & 1) * 16)));
        atomicAdd(&bins[be * 128 + col], hv);
        atomicAdd(&bins[binR[row] * 128 + col], hv);
      }
    }
  }
  __syncthreads();

  for (int i = tid; i < 27 * 128; i += 256)
    if (bins[i] != 0.f) atomicAdd(&agg[i], bins[i]);
  if (tid < 128) {
    atomicAdd(&sums[tid], lsum[tid]);
    atomicAdd(&sums[128 + tid], lsq[tid]);
  }
}

// ---------------------------------------------------------------------------
// Heads: per 64-row tile of households: h = relu(a3*z3+c3);
// for each head: t = relu(h @ W1T + b1) (WMMA, to LDS bf16);
//                out = t @ W2T + b2     (padded 16-col tiles, masked store)
// ---------------------------------------------------------------------------
__global__ __launch_bounds__(256) void k_heads(
    const uint16_t* __restrict__ zin, const float* __restrict__ ac3,
    const uint32_t* __restrict__ w1h, const uint32_t* __restrict__ w1e,
    const uint32_t* __restrict__ w1r, const uint32_t* __restrict__ w2h,
    const uint32_t* __restrict__ w2e, const uint32_t* __restrict__ w2r,
    const float* __restrict__ b1h, const float* __restrict__ b1e,
    const float* __restrict__ b1r, const float* __restrict__ b2h,
    const float* __restrict__ b2e, const float* __restrict__ b2r,
    float* __restrict__ out, int H, int dimH, int dimE, int dimR) {
  __shared__ uint32_t hT[64 * 64];    // 64 x 128 bf16
  __shared__ uint32_t tT[64 * 128];   // 64 x 256 bf16
  __shared__ float b1L[256];
  __shared__ float b2L[32];
  __shared__ float aL[128], cL[128];

  int tid = threadIdx.x;
  int r0  = blockIdx.x * 64;
  bool fullTile = (r0 + 64 <= H);
  if (tid < 128) { aL[tid] = ac3[tid]; cL[tid] = ac3[128 + tid]; }
  __syncthreads();

#ifdef USE_ASYNC_LDS
  for (int ch = tid; ch < 64 * 16; ch += 256) {
    int g = r0 + (ch >> 4);
    if (g < H)
      async_copy16((const char*)zin + (size_t)g * 256 + (size_t)(ch & 15) * 16,
                   (char*)hT + (size_t)ch * 16);
  }
  wait_async0();
  __syncthreads();
  for (int idx = tid; idx < 64 * 64; idx += 256) {
    int row = idx >> 6, cp = idx & 63, g = r0 + row;
    uint32_t u = 0;
    if (g < H) {
      uint32_t zp = hT[idx];
      int c0 = cp * 2;
      float h0 = fmaxf(fmaf(aL[c0],     b2f((uint16_t)zp),         cL[c0]),     0.f);
      float h1 = fmaxf(fmaf(aL[c0 + 1], b2f((uint16_t)(zp >> 16)), cL[c0 + 1]), 0.f);
      u = pack2(h0, h1);
    }
    hT[idx] = u;
  }
#else
  const uint32_t* zin32 = (const uint32_t*)zin;
  for (int idx = tid; idx < 64 * 64; idx += 256) {
    int row = idx >> 6, cp = idx & 63, g = r0 + row;
    uint32_t u = 0;
    if (g < H) {
      uint32_t zp = zin32[(size_t)g * 64 + cp];
      int c0 = cp * 2;
      float h0 = fmaxf(fmaf(aL[c0],     b2f((uint16_t)zp),         cL[c0]),     0.f);
      float h1 = fmaxf(fmaf(aL[c0 + 1], b2f((uint16_t)(zp >> 16)), cL[c0 + 1]), 0.f);
      u = pack2(h0, h1);
    }
    hT[idx] = u;
  }
#endif

  const uint32_t* w1s[3] = {w1h, w1e, w1r};
  const uint32_t* w2s[3] = {w2h, w2e, w2r};
  const float*    b1s[3] = {b1h, b1e, b1r};
  const float*    b2s[3] = {b2h, b2e, b2r};
  int    dims[3]  = {dimH, dimE, dimR};
  size_t bases[3] = {0, (size_t)H * dimH, (size_t)H * (dimH + dimE)};

  int w = tid >> 5, lane = tid & 31, hi = lane >> 4, nl = lane & 15;

#pragma unroll
  for (int hd = 0; hd < 3; ++hd) {
    if (tid < 256) b1L[tid] = b1s[hd][tid];
    if (tid < 32)  b2L[tid] = (tid < dims[hd]) ? b2s[hd][tid] : 0.f;
    __syncthreads();  // hT staged / previous head's tT consumed

    // GEMM1: 64x128 @ 128x256 -> t
    {
      int rb = w & 3, ng = w >> 2, m0 = rb * 16;
      ABFrag a4[4];
#pragma unroll
      for (int kc = 0; kc < 4; ++kc) a4[kc].v = ldA(hT, m0, kc, 64);
#pragma unroll
      for (int nn = 0; nn < 8; ++nn) {
        int nt = ng * 8 + nn;
        f32x8 acc = {};
#pragma unroll
        for (int kc = 0; kc < 4; ++kc) {
          bf16x16 b = ldB(w1s[hd], nt * 16, kc, 64);
          acc = wmma_bf16(a4[kc].v, b, acc);
        }
        int col = nt * 16 + nl;
        float bcol = b1L[col];
        uint16_t* tp = (uint16_t*)tT + (m0 + hi * 8) * 256 + col;
#pragma unroll
        for (int v = 0; v < 8; ++v) {
          float val = fmaxf(acc[v] + bcol, 0.f);
          tp[v * 256] = f2b(val);
        }
      }
    }
    __syncthreads();

    // GEMM2: 64x256 @ 256x{dim} -> out
    {
      int dim = dims[hd];
      int ctn = (dim + 15) / 16;
      int rb = w & 3, ct = w >> 2, m0 = rb * 16;
      if (ct < ctn) {
        f32x8 acc = {};
#pragma unroll
        for (int kc = 0; kc < 8; ++kc) {
          bf16x16 a = ldA(tT, m0, kc, 128);
          bf16x16 b = ldB(w2s[hd], ct * 16, kc, 128);
          acc = wmma_bf16(a, b, acc);
        }
        int col = ct * 16 + nl;
        if (col < dim) {
          float bb = b2L[col];
          float* op = out + bases[hd] + (size_t)(r0 + m0 + hi * 8) * dim + col;
          if (fullTile) {
#pragma unroll
            for (int v = 0; v < 8; ++v) op[(size_t)v * dim] = acc[v] + bb;
          } else {
#pragma unroll
            for (int v = 0; v < 8; ++v) {
              int g = r0 + m0 + v + hi * 8;
              if (g < H) op[(size_t)v * dim] = acc[v] + bb;
            }
          }
        }
      }
    }
    __syncthreads();
  }
}

// ---------------------------------------------------------------------------
// Host orchestration
// ---------------------------------------------------------------------------
extern "C" void kernel_launch(void* const* d_in, const int* in_sizes, int n_in,
                              void* d_out, int out_size, void* d_ws, size_t ws_size,
                              hipStream_t stream) {
  (void)n_in; (void)out_size; (void)ws_size;
  const float* x   = (const float*)d_in[0];
  const int*   ei  = (const int*)d_in[1];
  const float* w1l = (const float*)d_in[3];
  const float* w1r = (const float*)d_in[4];
  const float* b1  = (const float*)d_in[5];
  const float* w2l = (const float*)d_in[6];
  const float* w2r = (const float*)d_in[7];
  const float* b2  = (const float*)d_in[8];
  const float* w3l = (const float*)d_in[9];
  const float* w3r = (const float*)d_in[10];
  const float* b3  = (const float*)d_in[11];
  const float* g1w = (const float*)d_in[12], *g1b = (const float*)d_in[13], *g1m = (const float*)d_in[14];
  const float* g2w = (const float*)d_in[15], *g2b = (const float*)d_in[16], *g2m = (const float*)d_in[17];
  const float* g3w = (const float*)d_in[18], *g3b = (const float*)d_in[19], *g3m = (const float*)d_in[20];
  const float* hw1 = (const float*)d_in[21], *hb1 = (const float*)d_in[22];
  const float* hw2 = (const float*)d_in[23], *hb2 = (const float*)d_in[24];
  const float* ew1 = (const float*)d_in[25], *eb1 = (const float*)d_in[26];
  const float* ew2 = (const float*)d_in[27], *eb2 = (const float*)d_in[28];
  const float* rw1 = (const float*)d_in[29], *rb1 = (const float*)d_in[30];
  const float* rw2 = (const float*)d_in[31], *rb2 = (const float*)d_in[32];
  float* out = (float*)d_out;

  int N = in_sizes[0];         // total nodes (x is N x 1)
  int H = in_sizes[1] / 4;     // edge_index is 2 x 2H
  int dimH = in_sizes[24], dimE = in_sizes[28], dimR = in_sizes[32];
  const int* dst = ei + 2 * H;

  size_t off = 0;
  char* base = (char*)d_ws;
  auto alloc = [&](size_t bytes) -> void* {
    off = (off + 255) & ~(size_t)255;
    void* p = base + off;
    off += bytes;
    return p;
  };
  uint16_t* zA   = (uint16_t*)alloc((size_t)N * 128 * 2);
  uint16_t* zB   = (uint16_t*)alloc((size_t)N * 128 * 2);
  uint16_t* w2rT = (uint16_t*)alloc(128 * 128 * 2);
  uint16_t* w3rT = (uint16_t*)alloc(128 * 128 * 2);
  uint16_t* hw1T = (uint16_t*)alloc(256 * 128 * 2);
  uint16_t* ew1T = (uint16_t*)alloc(256 * 128 * 2);
  uint16_t* rw1T = (uint16_t*)alloc(256 * 128 * 2);
  uint16_t* hw2T = (uint16_t*)alloc(32 * 256 * 2);
  uint16_t* ew2T = (uint16_t*)alloc(32 * 256 * 2);
  uint16_t* rw2T = (uint16_t*)alloc(32 * 256 * 2);
  float* zreg  = (float*)alloc(0);  // start of zero-init region
  float* cnt   = (float*)alloc(32 * 4);
  float* agg1  = (float*)alloc(32 * 4);
  float* agg2  = (float*)alloc(27 * 128 * 4);
  float* agg3  = (float*)alloc(27 * 128 * 4);
  float* sums1 = (float*)alloc(256 * 4);
  float* sums2 = (float*)alloc(256 * 4);
  float* sums3 = (float*)alloc(256 * 4);
  size_t zbytes = (size_t)((base + off) - (char*)zreg);
  float* ac1 = (float*)alloc(256 * 4);
  float* ac2 = (float*)alloc(256 * 4);
  float* ac3 = (float*)alloc(256 * 4);

  (void)hipMemsetAsync(zreg, 0, zbytes, stream);

  // Weight conversion/transpose (bf16, [n][k])
  k_prepT<<<16, 256, 0, stream>>>(w2rT, w2r, 128, 128, 128);
  k_prepT<<<16, 256, 0, stream>>>(w3rT, w3r, 128, 128, 128);
  k_prepT<<<32, 256, 0, stream>>>(hw1T, hw1, 128, 256, 256);
  k_prepT<<<32, 256, 0, stream>>>(ew1T, ew1, 128, 256, 256);
  k_prepT<<<32, 256, 0, stream>>>(rw1T, rw1, 128, 256, 256);
  k_prepT<<<8, 256, 0, stream>>>(hw2T, hw2, 256, dimH, 32);
  k_prepT<<<8, 256, 0, stream>>>(ew2T, ew2, 256, dimE, 32);
  k_prepT<<<8, 256, 0, stream>>>(rw2T, rw2, 256, dimR, 32);

  // Edge counts + layer-1 scalar hub aggregation
  k_count_agg1<<<512, 256, 0, stream>>>(ei, x, cnt, agg1, H);

  // Layer 1 (outer product) + stats
  k_layer1<<<(N + 511) / 512, 128, 0, stream>>>(x, w1r, b1, zA, sums1, N);
  k_fixup<<<1, 256, 0, stream>>>(agg1, cnt, w1l, 1, zA, sums1, H);
  k_stats<<<1, 128, 0, stream>>>(sums1, g1w, g1b, g1m, ac1, 1.0f / (float)N);

  // Layer 2 (fused norm/relu + WMMA GEMM + stats + hub bins)
  k_ml<<<(N + 127) / 128, 256, 0, stream>>>(zA, ac1, (const uint32_t*)w2rT, b2, zB,
                                            sums2, agg2, dst, N, H);
  k_fixup<<<1, 256, 0, stream>>>(agg2, cnt, w2l, 128, zB, sums2, H);
  k_stats<<<1, 128, 0, stream>>>(sums2, g2w, g2b, g2m, ac2, 1.0f / (float)N);

  // Layer 3
  k_ml<<<(N + 127) / 128, 256, 0, stream>>>(zB, ac2, (const uint32_t*)w3rT, b3, zA,
                                            sums3, agg3, dst, N, H);
  k_fixup<<<1, 256, 0, stream>>>(agg3, cnt, w3l, 128, zA, sums3, H);
  k_stats<<<1, 128, 0, stream>>>(sums3, g3w, g3b, g3m, ac3, 1.0f / (float)N);

  // Heads (fused two-GEMM MLPs into concatenated d_out)
  k_heads<<<(H + 63) / 64, 256, 0, stream>>>(
      zA, ac3, (const uint32_t*)hw1T, (const uint32_t*)ew1T, (const uint32_t*)rw1T,
      (const uint32_t*)hw2T, (const uint32_t*)ew2T, (const uint32_t*)rw2T,
      hb1, eb1, rb1, hb2, eb2, rb2, out, H, dimH, dimE, dimR);
}
